// GridMLP_88407606821461
// MI455X (gfx1250) — matrix operand order repaired and checked by reading, not compiled, not run on hardware
//
#include <hip/hip_runtime.h>

typedef __attribute__((ext_vector_type(16))) _Float16 v16h;
typedef __attribute__((ext_vector_type(8)))  _Float16 v8h;
typedef __attribute__((ext_vector_type(8)))  float    v8f;

// ---------------------------------------------------------------------------
// Weight prep: repack W[din][dout] (f32) into per-lane WMMA B-matrix layout
// (f16), padded. Block b = 512 halfs. Layer block bases: L0:0(12) L1:12(8)
// L2:20(8) L3:28(2).  B layout (16x16x32 f16, wave32):
//   lane<16 : col = lane,    K = kc*32 + j (j<8), kc*32 + 16 + (j-8) (j>=8)
//   lane>=16: col = lane-16, K offset +8 (same split)
// ---------------------------------------------------------------------------
__global__ void prep_weights(const float* __restrict__ W0, const float* __restrict__ W1,
                             const float* __restrict__ W2, const float* __restrict__ W3,
                             _Float16* __restrict__ WB) {
    int idx = blockIdx.x * blockDim.x + threadIdx.x;
    if (idx >= 30 * 512) return;
    int b    = idx >> 9;
    int e    = idx & 511;
    int lane = e >> 4;
    int j    = e & 15;
    int base, KC, din, dout;
    const float* W;
    if (b < 12)      { base = 0;  KC = 3; din = 90; dout = 64; W = W0; }
    else if (b < 20) { base = 12; KC = 2; din = 64; dout = 64; W = W1; }
    else if (b < 28) { base = 20; KC = 2; din = 64; dout = 64; W = W2; }
    else             { base = 28; KC = 2; din = 64; dout = 3;  W = W3; }
    int local = b - base;
    int nt = local / KC;
    int kc = local % KC;
    int n  = nt * 16 + (lane & 15);
    int k  = kc * 32 + ((lane & 16) >> 1) + (j & 7) + ((j & 8) << 1);
    float v = (k < din && n < dout) ? W[k * dout + n] : 0.0f;
    WB[idx] = (_Float16)v;
}

// ---------------------------------------------------------------------------
// Helpers
// ---------------------------------------------------------------------------
__device__ __forceinline__ void st_pair(_Float16* p, float a, float b) {
    union { _Float16 h[2]; unsigned u; } u2;
    u2.h[0] = (_Float16)a; u2.h[1] = (_Float16)b;
    *(unsigned*)p = u2.u;                       // ds_store_b32
}

// bilinear grid sample, align_corners=True, border clamp; 16 channels -> rowp[0..15]
__device__ __forceinline__ void sample_level(const float* __restrict__ g, int R,
                                             float x, float y, _Float16* rowp) {
    float s  = 0.5f * (float)(R - 1);
    float fx = fminf(fmaxf((x + 1.0f) * s, 0.0f), (float)(R - 1));
    float fy = fminf(fmaxf((y + 1.0f) * s, 0.0f), (float)(R - 1));
    float x0f = floorf(fx), y0f = floorf(fy);
    float wx = fx - x0f, wy = fy - y0f;
    int x0 = (int)x0f, y0 = (int)y0f;
    int x1 = min(x0 + 1, R - 1), y1 = min(y0 + 1, R - 1);
    int i00 = y0 * R + x0, i01 = y0 * R + x1;
    int i10 = y1 * R + x0, i11 = y1 * R + x1;
    float w00 = (1.0f - wx) * (1.0f - wy), w01 = wx * (1.0f - wy);
    float w10 = (1.0f - wx) * wy,          w11 = wx * wy;
    int RR = R * R;
#pragma unroll
    for (int c = 0; c < 16; c += 2) {
        const float* gc0 = g + c * RR;
        const float* gc1 = gc0 + RR;
        float v0 = gc0[i00] * w00 + gc0[i01] * w01 + gc0[i10] * w10 + gc0[i11] * w11;
        float v1 = gc1[i00] * w00 + gc1[i01] * w01 + gc1[i10] * w10 + gc1[i11] * w11;
        st_pair(rowp + c, v0, v1);
    }
}

// One hidden layer (dout=64, ReLU): hin [32][sIn] f16 -> hout [32][64] f16.
// A layout (16x16x32 f16): lane<16 row=lane, elems 0..7=K+0..7, 8..15=K+16..23;
// lane>=16 row=lane-16, K offset +8 / +24.
template <int KC>
__device__ __forceinline__ void mlp_hidden_layer(const _Float16* __restrict__ WB, int blockBase,
                                                 const _Float16* hin, int sIn,
                                                 _Float16* hout,
                                                 const float* __restrict__ bias, int lane) {
    int r  = lane & 15;
    int hi = (lane >> 4) & 1;
#pragma unroll
    for (int mt = 0; mt < 2; ++mt) {
        v16h A[KC];
#pragma unroll
        for (int kc = 0; kc < KC; ++kc) {
            int row = mt * 16 + r;
            int kb  = kc * 32 + hi * 8;
            v8h lo = *(const v8h*)(hin + row * sIn + kb);
            v8h hh = *(const v8h*)(hin + row * sIn + kb + 16);
#pragma unroll
            for (int i = 0; i < 8; ++i) { A[kc][i] = lo[i]; A[kc][8 + i] = hh[i]; }
        }
#pragma unroll
        for (int nt = 0; nt < 4; ++nt) {
            float bv = bias[nt * 16 + r];
            v8f acc;
#pragma unroll
            for (int i = 0; i < 8; ++i) acc[i] = bv;
#pragma unroll
            for (int kc = 0; kc < KC; ++kc) {
                const _Float16* bp = WB + (size_t)(blockBase + nt * KC + kc) * 512 + lane * 16;
                v16h B = *(const v16h*)bp;
                acc = __builtin_amdgcn_wmma_f32_16x16x32_f16(false, A[kc], false, B,
                                                             (short)0, acc, false, false);
            }
#pragma unroll
            for (int i = 0; i < 8; ++i) {
                float vv = fmaxf(acc[i], 0.0f);
                hout[(mt * 16 + hi * 8 + i) * 64 + nt * 16 + r] = (_Float16)vv;
            }
        }
    }
}

// ---------------------------------------------------------------------------
// Fused grid-sample + encode + 4-layer MLP. 128 threads = 4 waves, 32 pts/wave.
// ---------------------------------------------------------------------------
__global__ void __launch_bounds__(128)
gridmlp_kernel(const float* __restrict__ xy, const float* __restrict__ tfeat,
               const float* __restrict__ g0, const float* __restrict__ g1,
               const float* __restrict__ g2, const float* __restrict__ g3,
               const float* __restrict__ b0, const float* __restrict__ b1,
               const float* __restrict__ b2, const float* __restrict__ b3,
               const _Float16* __restrict__ WB, float* __restrict__ out, int N) {
    __shared__ __align__(16) _Float16 sA[4][32][96];   // h0 (96-wide); reused for h2 (64-wide)
    __shared__ __align__(16) _Float16 sB[4][32][64];   // h1; reused for h3

    int tid   = threadIdx.x;
    int w     = tid >> 5;
    int lane  = tid & 31;
    int pbase = blockIdx.x * 128 + w * 32;
    int p     = pbase + lane;
    int pc    = (p < N) ? p : (N - 1);

    // ---- per-lane feature build: one point per lane, f16 row in LDS ----
    _Float16* row = &sA[w][lane][0];
    float x = xy[2 * (size_t)pc + 0];
    float y = xy[2 * (size_t)pc + 1];
    sample_level(g0, 16,  x, y, row + 0);
    sample_level(g1, 32,  x, y, row + 16);
    sample_level(g2, 64,  x, y, row + 32);
    sample_level(g3, 128, x, y, row + 48);

    const float PI = 3.14159265358979323846f;
    st_pair(row + 64, x, y);
    st_pair(row + 66, __sinf(PI * x),        __sinf(PI * y));
    st_pair(row + 68, __cosf(PI * x),        __cosf(PI * y));
    st_pair(row + 70, __sinf(2.0f * PI * x), __sinf(2.0f * PI * y));
    st_pair(row + 72, __cosf(2.0f * PI * x), __cosf(2.0f * PI * y));

    const float4* tf = (const float4*)(tfeat + 16 * (size_t)pc);
#pragma unroll
    for (int q = 0; q < 4; ++q) {
        float4 v = tf[q];
        st_pair(row + 74 + q * 4 + 0, v.x, v.y);
        st_pair(row + 74 + q * 4 + 2, v.z, v.w);
    }
    st_pair(row + 90, 0.0f, 0.0f);   // K padding 90..95
    st_pair(row + 92, 0.0f, 0.0f);
    st_pair(row + 94, 0.0f, 0.0f);

    // ---- MLP: WMMA tiles, wave-private LDS ping-pong ----
    mlp_hidden_layer<3>(WB, 0,  &sA[w][0][0], 96, &sB[w][0][0], b0, lane);  // h0 -> h1
    mlp_hidden_layer<2>(WB, 12, &sB[w][0][0], 64, &sA[w][0][0], b1, lane);  // h1 -> h2
    mlp_hidden_layer<2>(WB, 20, &sA[w][0][0], 64, &sB[w][0][0], b2, lane);  // h2 -> h3

    // ---- output layer: 64 -> 3 (padded to 16 cols), f32 store ----
    {
        int r  = lane & 15;
        int hi = (lane >> 4) & 1;
#pragma unroll
        for (int mt = 0; mt < 2; ++mt) {
            v16h A[2];
#pragma unroll
            for (int kc = 0; kc < 2; ++kc) {
                int rrow = mt * 16 + r;
                int kb   = kc * 32 + hi * 8;
                v8h lo = *(const v8h*)(&sB[w][rrow][kb]);
                v8h hh = *(const v8h*)(&sB[w][rrow][kb + 16]);
#pragma unroll
                for (int i = 0; i < 8; ++i) { A[kc][i] = lo[i]; A[kc][8 + i] = hh[i]; }
            }
            float bv = (r < 3) ? b3[r] : 0.0f;
            v8f acc;
#pragma unroll
            for (int i = 0; i < 8; ++i) acc[i] = bv;
#pragma unroll
            for (int kc = 0; kc < 2; ++kc) {
                const _Float16* bp = WB + (size_t)(28 + kc) * 512 + lane * 16;
                v16h B = *(const v16h*)bp;
                acc = __builtin_amdgcn_wmma_f32_16x16x32_f16(false, A[kc], false, B,
                                                             (short)0, acc, false, false);
            }
            if (r < 3) {
#pragma unroll
                for (int i = 0; i < 8; ++i) {
                    int pt = pbase + mt * 16 + hi * 8 + i;
                    if (pt < N) out[(size_t)pt * 3 + r] = acc[i];
                }
            }
        }
    }
}

// ---------------------------------------------------------------------------
extern "C" void kernel_launch(void* const* d_in, const int* in_sizes, int n_in,
                              void* d_out, int out_size, void* d_ws, size_t ws_size,
                              hipStream_t stream) {
    const float* xy = (const float*)d_in[0];
    const float* tf = (const float*)d_in[1];
    const float* g0 = (const float*)d_in[2];
    const float* g1 = (const float*)d_in[3];
    const float* g2 = (const float*)d_in[4];
    const float* g3 = (const float*)d_in[5];
    const float* W0 = (const float*)d_in[6];
    const float* b0 = (const float*)d_in[7];
    const float* W1 = (const float*)d_in[8];
    const float* b1 = (const float*)d_in[9];
    const float* W2 = (const float*)d_in[10];
    const float* b2 = (const float*)d_in[11];
    const float* W3 = (const float*)d_in[12];
    const float* b3 = (const float*)d_in[13];
    _Float16* WB = (_Float16*)d_ws;            // 30 * 512 halfs = 30720 B
    float* out = (float*)d_out;
    int N = in_sizes[0] / 2;

    prep_weights<<<(30 * 512 + 255) / 256, 256, 0, stream>>>(W0, W1, W2, W3, WB);
    int nblocks = (N + 127) / 128;
    gridmlp_kernel<<<nblocks, 128, 0, stream>>>(xy, tf, g0, g1, g2, g3,
                                                b0, b1, b2, b3, WB, out, N);
}
